// SimpleGCN_10660108829072
// MI455X (gfx1250) — compile-verified, matrix operand order checked by reading, not run on hardware
//
#include <hip/hip_runtime.h>
#include <hip/hip_bf16.h>

typedef __attribute__((ext_vector_type(16))) _Float16 v16h;
typedef __attribute__((ext_vector_type(8)))  float    v8f;

// ---------------------------------------------------------------------------
// Degree kernels: deg[i] = 1 (self loop) + #edges with col==i ; then rsqrt.
// ---------------------------------------------------------------------------
__global__ void deg_init(float* __restrict__ deg, int n) {
    int i = blockIdx.x * blockDim.x + threadIdx.x;
    if (i < n) deg[i] = 1.0f;
}

__global__ void deg_count(const long long* __restrict__ col, float* __restrict__ deg,
                          long long nedges) {
    long long i = (long long)blockIdx.x * blockDim.x + threadIdx.x;
    long long stride = (long long)gridDim.x * blockDim.x;
    for (; i < nedges; i += stride)
        atomicAdd(&deg[(int)col[i]], 1.0f);
}

__global__ void deg_rsqrt(float* __restrict__ deg, int n) {
    int i = blockIdx.x * blockDim.x + threadIdx.x;
    if (i < n) deg[i] = rsqrtf(deg[i]);
}

// ---------------------------------------------------------------------------
// WMMA GEMM:  Y[nrows x NOUT] = f(X)[nrows x K] @ W[K x NOUT]
// f(X) = RELU_IN ? relu(X + inBias) : X   (fuses previous layer's bias+ReLU)
// Block = 256 threads = 8 wave32; each wave computes a 16 x NOUT tile.
// W staged in LDS transposed as f16: wT[n][k], so B fragments are two
// contiguous 16B spans per lane. A loaded from global f32 -> f16.
// CDNA5 16-bit A/B lane layout: lanes 0-15 hold K {0..7,16..23}+k0,
// lanes 16-31 hold K {8..15,24..31}+k0. C/D: VGPR i, lanes 0-15 -> M=i,
// lanes 16-31 -> M=8+i, N = lane&15.
// ---------------------------------------------------------------------------
template<int K, int NOUT, bool RELU_IN>
__global__ __launch_bounds__(256)
void gemm_wmma(const float* __restrict__ X, const float* __restrict__ W,
               const float* __restrict__ inBias, float* __restrict__ Y, int nrows)
{
    __shared__ _Float16 wT[NOUT * K];
    for (int idx = threadIdx.x; idx < NOUT * K; idx += 256) {
        int n = idx / K;
        int k = idx - n * K;
        wT[n * K + k] = (_Float16)W[k * NOUT + n];
    }
    __syncthreads();

    const int lane  = threadIdx.x & 31;
    const int wave  = threadIdx.x >> 5;
    const int m0    = blockIdx.x * 128 + wave * 16;
    const int mrow  = m0 + (lane & 15);
    const int mcl   = (mrow < nrows) ? mrow : (nrows - 1);  // clamp, keep EXEC full
    const int kbase = (lane < 16) ? 0 : 8;

    constexpr int NT = NOUT / 16;
    v8f acc[NT];
    const v8f vzero = {};
#pragma unroll
    for (int t = 0; t < NT; ++t) acc[t] = vzero;

#pragma unroll
    for (int kk = 0; kk < K; kk += 32) {
        // ---- A fragment (16x32 f16) ----
        const float* xp = X + (long long)mcl * K + kk + kbase;
        v16h a;
#pragma unroll
        for (int e = 0; e < 8; ++e) {
            float lo = xp[e];
            float hi = xp[e + 16];
            if (RELU_IN) {
                lo = fmaxf(lo + inBias[kk + kbase + e], 0.0f);
                hi = fmaxf(hi + inBias[kk + kbase + 16 + e], 0.0f);
            }
            a[e]     = (_Float16)lo;
            a[e + 8] = (_Float16)hi;
        }
        // ---- one WMMA per 16-col output tile ----
#pragma unroll
        for (int t = 0; t < NT; ++t) {
            const _Float16* wp = &wT[(t * 16 + (lane & 15)) * K + kk + kbase];
            v16h b;
#pragma unroll
            for (int e = 0; e < 8; ++e) {
                b[e]     = wp[e];
                b[e + 8] = wp[e + 16];
            }
            acc[t] = __builtin_amdgcn_wmma_f32_16x16x32_f16(
                false, a, false, b, (short)0, acc[t], false, false);
        }
    }

    const int rowoff = (lane >> 4) * 8;
    const int ncol   = lane & 15;
#pragma unroll
    for (int i = 0; i < 8; ++i) {
        int m = m0 + rowoff + i;
        if (m < nrows) {
#pragma unroll
            for (int t = 0; t < NT; ++t)
                Y[(long long)m * NOUT + t * 16 + ncol] = acc[t][i];
        }
    }
}

// ---------------------------------------------------------------------------
// Aggregation: A = D^{-1/2}(Adj+I)D^{-1/2} H
// Step 1 (init, also handles self loop): A[i,:] = H[i,:] * dinv[i]^2
// Step 2 (edges): A[c,:] += H[r,:] * dinv[r]*dinv[c]  via float atomics,
//                 F/4 lanes per edge, float4 gathers.
// ---------------------------------------------------------------------------
template<int F>
__global__ void agg_self_init(const float* __restrict__ H, const float* __restrict__ dinv,
                              float* __restrict__ A, int nnodes)
{
    long long total  = (long long)nnodes * F;
    long long stride = (long long)gridDim.x * blockDim.x;
    for (long long i = (long long)blockIdx.x * blockDim.x + threadIdx.x; i < total; i += stride) {
        int node = (int)(i / F);
        float w = dinv[node];
        A[i] = H[i] * w * w;
    }
}

template<int F>
__global__ void agg_edges(const float* __restrict__ H, const float* __restrict__ dinv,
                          const long long* __restrict__ row, const long long* __restrict__ col,
                          float* __restrict__ A, long long nedges)
{
    constexpr int C = F / 4;                 // float4 chunks per edge
    long long total  = nedges * C;
    long long stride = (long long)gridDim.x * blockDim.x;
    for (long long t = (long long)blockIdx.x * blockDim.x + threadIdx.x; t < total; t += stride) {
        long long e = t / C;
        int       j = (int)(t - e * C);
        int r = (int)row[e];
        int c = (int)col[e];
        float w = dinv[r] * dinv[c];
        float4 h4 = ((const float4*)(H + (long long)r * F))[j];
        float* dst = A + (long long)c * F + j * 4;
        atomicAdd(dst + 0, h4.x * w);
        atomicAdd(dst + 1, h4.y * w);
        atomicAdd(dst + 2, h4.z * w);
        atomicAdd(dst + 3, h4.w * w);
    }
}

// ---------------------------------------------------------------------------
// Final projection: out[i] = relu(A2[i,:] + b2) . Wout + bout   (32 -> 1)
// ---------------------------------------------------------------------------
__global__ void final_out(const float* __restrict__ A2, const float* __restrict__ b2,
                          const float* __restrict__ Wout, const float* __restrict__ bout,
                          float* __restrict__ out, int n)
{
    int i = blockIdx.x * blockDim.x + threadIdx.x;
    if (i < n) {
        const float* a = A2 + (long long)i * 32;
        float s = bout[0];
#pragma unroll
        for (int f = 0; f < 32; ++f)
            s += fmaxf(a[f] + b2[f], 0.0f) * Wout[f];
        out[i] = s;
    }
}

// ---------------------------------------------------------------------------
extern "C" void kernel_launch(void* const* d_in, const int* in_sizes, int n_in,
                              void* d_out, int out_size, void* d_ws, size_t ws_size,
                              hipStream_t stream)
{
    const float*     x    = (const float*)d_in[0];
    const long long* ei   = (const long long*)d_in[1];
    const float*     W1   = (const float*)d_in[2];
    const float*     b1   = (const float*)d_in[3];
    const float*     W2   = (const float*)d_in[4];
    const float*     b2   = (const float*)d_in[5];
    const float*     Wout = (const float*)d_in[6];
    const float*     bout = (const float*)d_in[7];
    float*           out  = (float*)d_out;

    const int       n = in_sizes[0] / 256;     // 100000 nodes
    const long long E = in_sizes[1] / 2;       // 3.2M edges
    const long long* row = ei;
    const long long* col = ei + E;

    // Workspace layout (aliased to keep peak ~51 MB):
    //   dinv : n floats (padded)
    //   bufA : n*64 floats  -> h1, then reused as h2 (first n*32) + a2 (second n*32)
    //   bufB : n*64 floats  -> a1
    float* dinv = (float*)d_ws;
    size_t pad  = ((size_t)n + 255) & ~(size_t)255;
    float* bufA = dinv + pad;
    float* bufB = bufA + (size_t)n * 64;
    float* h1 = bufA;
    float* a1 = bufB;
    float* h2 = bufA;                    // overwrites h1 (dead after agg1)
    float* a2 = bufA + (size_t)n * 32;   // disjoint from h2 region

    // --- degrees ---
    deg_init<<<(n + 255) / 256, 256, 0, stream>>>(dinv, n);
    deg_count<<<2048, 256, 0, stream>>>(col, dinv, E);
    deg_rsqrt<<<(n + 255) / 256, 256, 0, stream>>>(dinv, n);

    const int gemm_blocks = (n + 127) / 128;

    // --- layer 1: h1 = x @ W1 ; a1 = norm-aggregate(h1) ---
    gemm_wmma<256, 64, false><<<gemm_blocks, 256, 0, stream>>>(x, W1, nullptr, h1, n);
    agg_self_init<64><<<4096, 256, 0, stream>>>(h1, dinv, a1, n);
    agg_edges<64><<<8192, 256, 0, stream>>>(h1, dinv, row, col, a1, E);

    // --- layer 2: h2 = relu(a1 + b1) @ W2 ; a2 = norm-aggregate(h2) ---
    gemm_wmma<64, 32, true><<<gemm_blocks, 256, 0, stream>>>(a1, W2, b1, h2, n);
    agg_self_init<32><<<4096, 256, 0, stream>>>(h2, dinv, a2, n);
    agg_edges<32><<<8192, 256, 0, stream>>>(h2, dinv, row, col, a2, E);

    // --- output head: out = relu(a2 + b2) @ Wout + bout ---
    final_out<<<(n + 255) / 256, 256, 0, stream>>>(a2, b2, Wout, bout, out, n);
}
